// MultiHeadAttention_78675210928269
// MI455X (gfx1250) — matrix operand order repaired and checked by reading, not compiled, not run on hardware
//
#include <hip/hip_runtime.h>
#include <math.h>

// ---------------------------------------------------------------------------
// MI455X (gfx1250) fused attention block: LN -> QKV GEMM -> QK-LN -> RoPE ->
// masked flash attention -> out GEMM.  All matmuls on v_wmma_f32_16x16x32_bf16.
// wave32 everywhere. GEMM uses double-buffered LDS tiles with async global->LDS
// copies (ASYNCcnt) when the toolchain exposes them.
// ---------------------------------------------------------------------------

#define B_    2
#define L_    2048
#define D_    1024
#define H_    16
#define DH_   64
#define ROWS_ (B_ * L_)   // 4096
#define QKVN_ (3 * D_)    // 3072

typedef __bf16 bf16;
typedef __attribute__((ext_vector_type(16))) __bf16 v16bf;
typedef __attribute__((ext_vector_type(8)))  float  v8f;

union FragB { uint4 u[2]; v16bf v; };
static_assert(sizeof(v16bf) == 32, "v16bf must be 32 bytes");
static_assert(sizeof(FragB) == 32, "FragB must be 32 bytes");

#if defined(__AMDGCN__) && \
    __has_builtin(__builtin_amdgcn_global_load_async_to_lds_b128) && \
    __has_builtin(__builtin_amdgcn_s_wait_asynccnt)
#define USE_ASYNC_LDS 1
#else
#define USE_ASYNC_LDS 0
#endif

#if USE_ASYNC_LDS
typedef int v4i_pay __attribute__((vector_size(16)));   // 128-bit payload type
// one 16-byte lane-granular async copy, global -> LDS (tracked by ASYNCcnt)
__device__ __forceinline__ void async_copy16(const bf16* g, bf16* l) {
  __builtin_amdgcn_global_load_async_to_lds_b128(
      (v4i_pay*)g, (v4i_pay*)l, 0, 0);
}
#endif

// float -> bf16 with round-to-nearest-even (avoid relying on __bf16 casts)
__device__ __forceinline__ bf16 f2bf(float f) {
  union { float f; unsigned u; } a; a.f = f;
  unsigned r = (a.u + 0x7FFFu + ((a.u >> 16) & 1u)) >> 16;
  union { unsigned short s; bf16 b; } o; o.s = (unsigned short)r;
  return o.b;
}

// Block (256 threads = 8 waves) sum reduction; sb must hold >= 9 floats.
__device__ __forceinline__ float block_reduce_sum(float v, float* sb) {
  #pragma unroll
  for (int o = 16; o; o >>= 1) v += __shfl_xor(v, o, 32);
  int w = threadIdx.x >> 5, ln = threadIdx.x & 31;
  __syncthreads();
  if (ln == 0) sb[w] = v;
  __syncthreads();
  if (threadIdx.x == 0) {
    float t = 0.f;
    #pragma unroll
    for (int i = 0; i < 8; i++) t += sb[i];
    sb[8] = t;
  }
  __syncthreads();
  return sb[8];
}

// ---------------------------------------------------------------------------
// 1) Weight convert f32 -> bf16 with transpose: W[K][N] -> Wt[N][K]
// ---------------------------------------------------------------------------
__global__ void cvt_transpose_kernel(const float* __restrict__ W,
                                     bf16* __restrict__ Wt, int K, int N) {
  int idx = blockIdx.x * blockDim.x + threadIdx.x;
  if (idx >= K * N) return;
  int k = idx / N, n = idx % N;
  Wt[(long)n * K + k] = f2bf(W[idx]);
}

// ---------------------------------------------------------------------------
// 2) LN1: h = LN(x) * w + b   (f32 in, bf16 out), one block per row of 1024
// ---------------------------------------------------------------------------
__global__ __launch_bounds__(256)
void ln1_kernel(const float* __restrict__ x, const float* __restrict__ w,
                const float* __restrict__ bias, bf16* __restrict__ h) {
  __shared__ float sb[9];
  const int row = blockIdx.x;
  const float* xr = x + (long)row * D_;
  float v[4], s = 0.f, s2 = 0.f;
  #pragma unroll
  for (int i = 0; i < 4; i++) {
    int c = threadIdx.x + 256 * i;
    v[i] = xr[c]; s += v[i]; s2 += v[i] * v[i];
  }
  float sum  = block_reduce_sum(s,  sb);
  float sum2 = block_reduce_sum(s2, sb);
  float mu  = sum  * (1.f / D_);
  float var = sum2 * (1.f / D_) - mu * mu;
  float rs  = rsqrtf(var + 1e-5f);
  #pragma unroll
  for (int i = 0; i < 4; i++) {
    int c = threadIdx.x + 256 * i;
    h[(long)row * D_ + c] = f2bf((v[i] - mu) * rs * w[c] + bias[c]);
  }
}

// ---------------------------------------------------------------------------
// 3) Tiled bf16 WMMA GEMM:  C[M][N] = A[M][K] * Wt[N][K]^T   (C in f32)
//    Block tile 128x128, 8 waves of 32x64, K step 32.
//    Double-buffered LDS; async global->LDS copies overlap the WMMAs.
// ---------------------------------------------------------------------------
__global__ __launch_bounds__(256)
void gemm_bf16_wmma_kernel(const bf16* __restrict__ A,
                           const bf16* __restrict__ Wt,
                           float* __restrict__ C, int M, int N, int K) {
  __shared__ __align__(16) bf16 lds_a[2][128 * 32];
  __shared__ __align__(16) bf16 lds_w[2][128 * 32];
  const int m0 = blockIdx.y * 128;
  const int n0 = blockIdx.x * 128;
  const int tid  = threadIdx.x;
  const int wave = tid >> 5, lane = tid & 31;
  const int half = lane >> 4, nn = lane & 15;
  const int wm = (wave & 3) * 32, wn = (wave >> 2) * 64;
  const int r = tid >> 1, seg = (tid & 1) * 16;   // 2 threads per tile row

  const bf16* gA = A  + (long)(m0 + r) * K + seg;
  const bf16* gW = Wt + (long)(n0 + r) * K + seg;
  bf16* laBase = &lds_a[0][r * 32 + seg];
  bf16* lwBase = &lds_w[0][r * 32 + seg];
  const int bufStride = 128 * 32;

  v8f acc[2][4] = {};

  // ---- prologue: fetch tile 0 into buffer 0 --------------------------------
#if USE_ASYNC_LDS
  async_copy16(gA + 0, laBase + 0);
  async_copy16(gA + 8, laBase + 8);
  async_copy16(gW + 0, lwBase + 0);
  async_copy16(gW + 8, lwBase + 8);
  __builtin_amdgcn_s_wait_asynccnt(0);
#else
  {
    const uint4* pa = (const uint4*)gA;
    const uint4* pw = (const uint4*)gW;
    uint4* la = (uint4*)laBase;
    uint4* lw = (uint4*)lwBase;
    la[0] = pa[0]; la[1] = pa[1];
    lw[0] = pw[0]; lw[1] = pw[1];
  }
#endif
  __syncthreads();

  int buf = 0;
  for (int k0 = 0; k0 < K; k0 += 32) {
    // ---- issue next tile into the other buffer (overlaps WMMAs below) -----
    const int nxt = buf ^ 1;
    if (k0 + 32 < K) {
#if USE_ASYNC_LDS
      async_copy16(gA + k0 + 32 + 0, laBase + nxt * bufStride + 0);
      async_copy16(gA + k0 + 32 + 8, laBase + nxt * bufStride + 8);
      async_copy16(gW + k0 + 32 + 0, lwBase + nxt * bufStride + 0);
      async_copy16(gW + k0 + 32 + 8, lwBase + nxt * bufStride + 8);
#else
      __builtin_prefetch(gA + k0 + 32, 0, 1);
      __builtin_prefetch(gW + k0 + 32, 0, 1);
      const uint4* pa = (const uint4*)(gA + k0 + 32);
      const uint4* pw = (const uint4*)(gW + k0 + 32);
      uint4* la = (uint4*)(laBase + nxt * bufStride);
      uint4* lw = (uint4*)(lwBase + nxt * bufStride);
      la[0] = pa[0]; la[1] = pa[1];
      lw[0] = pw[0]; lw[1] = pw[1];
#endif
    }

    // ---- compute current buffer -------------------------------------------
    const bf16* cur_a = &lds_a[0][0] + buf * bufStride;
    const bf16* cur_w = &lds_w[0][0] + buf * bufStride;
    FragB fa[2], fb[4];
    #pragma unroll
    for (int i = 0; i < 2; i++) {     // A 16x32: half -> K {0..15}/{16..31}
      const bf16* p = cur_a + (wm + i * 16 + nn) * 32;
      fa[i].u[0] = *(const uint4*)(p + 16 * half);
      fa[i].u[1] = *(const uint4*)(p + 16 * half + 8);
    }
    #pragma unroll
    for (int j = 0; j < 4; j++) {     // B 32x16: half -> K {0..15}/{16..31}
      const bf16* p = cur_w + (wn + j * 16 + nn) * 32;
      fb[j].u[0] = *(const uint4*)(p + 16 * half);
      fb[j].u[1] = *(const uint4*)(p + 16 * half + 8);
    }
    #pragma unroll
    for (int i = 0; i < 2; i++)
      #pragma unroll
      for (int j = 0; j < 4; j++)
        acc[i][j] = __builtin_amdgcn_wmma_f32_16x16x32_bf16(
            false, fa[i].v, false, fb[j].v, (short)0, acc[i][j], false, false);

    // ---- publish: my async/staged copies done + everyone done reading -----
#if USE_ASYNC_LDS
    __builtin_amdgcn_s_wait_asynccnt(0);
#endif
    __syncthreads();
    buf = nxt;
  }

  #pragma unroll
  for (int i = 0; i < 2; i++)
    #pragma unroll
    for (int j = 0; j < 4; j++)
      #pragma unroll
      for (int rr = 0; rr < 8; rr++) {
        int row = m0 + wm + i * 16 + 8 * half + rr;   // C layout: VGPR rr -> row
        int col = n0 + wn + j * 16 + nn;
        C[(long)row * N + col] = acc[i][j][rr];
      }
}

// ---------------------------------------------------------------------------
// 4) Q/K layernorm (over full D) + RoPE + scatter to per-head layouts.
//    q,k -> [B][H][L][64] bf16 ; v -> transposed [B][H][64][L] bf16
// ---------------------------------------------------------------------------
__global__ __launch_bounds__(256)
void qk_ln_rope_kernel(const float* __restrict__ qkv,
                       const float* __restrict__ qw, const float* __restrict__ kw,
                       bf16* __restrict__ qo, bf16* __restrict__ ko,
                       bf16* __restrict__ vto) {
  __shared__ float sb[9];
  __shared__ float sq[D_];
  __shared__ float sk[D_];
  const int row = blockIdx.x, b = row >> 11, l = row & (L_ - 1);
  const float* base = qkv + (long)row * QKVN_;
  float qv[4], kv[4];
  float s_q = 0.f, s2_q = 0.f, s_k = 0.f, s2_k = 0.f;
  #pragma unroll
  for (int i = 0; i < 4; i++) {
    int c = threadIdx.x + 256 * i;
    qv[i] = base[c]; kv[i] = base[D_ + c];
    s_q += qv[i]; s2_q += qv[i] * qv[i];
    s_k += kv[i]; s2_k += kv[i] * kv[i];
  }
  float sumq  = block_reduce_sum(s_q,  sb);
  float sumq2 = block_reduce_sum(s2_q, sb);
  float sumk  = block_reduce_sum(s_k,  sb);
  float sumk2 = block_reduce_sum(s2_k, sb);
  float muq = sumq * (1.f / D_), vq = sumq2 * (1.f / D_) - muq * muq;
  float muk = sumk * (1.f / D_), vk = sumk2 * (1.f / D_) - muk * muk;
  float rq = rsqrtf(vq + 1e-5f), rk = rsqrtf(vk + 1e-5f);
  #pragma unroll
  for (int i = 0; i < 4; i++) {
    int c = threadIdx.x + 256 * i;
    sq[c] = (qv[i] - muq) * rq * qw[c];
    sk[c] = (kv[i] - muk) * rk * kw[c];
  }
  __syncthreads();
  #pragma unroll
  for (int i = 0; i < 4; i++) {
    int c = threadIdx.x + 256 * i;
    int hh = c >> 6, d = c & 63;
    int dr = d & 31;                       // concat(freqs,freqs): period 32
    float freq = __powf(10000.0f, -(float)(2 * dr) * (1.0f / 64.0f));
    float ang = (float)l * freq;
    float sn, cs; __sincosf(ang, &sn, &cs);
    int   partner = (d < 32) ? c + 32 : c - 32;
    float sgn     = (d < 32) ? -1.f : 1.f;
    float qr_ = sq[c] * cs + sgn * sq[partner] * sn;
    float kr_ = sk[c] * cs + sgn * sk[partner] * sn;
    long qk_idx = ((long)(b * H_ + hh) * L_ + l) * DH_ + d;
    qo[qk_idx] = f2bf(qr_);
    ko[qk_idx] = f2bf(kr_);
    vto[((long)(b * H_ + hh) * DH_ + d) * L_ + l] = f2bf(base[2 * D_ + c]);
  }
}

// ---------------------------------------------------------------------------
// 5) Flash attention with segment-id mask. One wave per 32-row Q tile
//    (two 16-row sub-tiles) so every K/V fragment feeds TWO WMMAs.
//    scores: 8 WMMA / 32-kv chunk; ctx: 8 WMMA / chunk.
// ---------------------------------------------------------------------------
__global__ __launch_bounds__(256)
void attn_kernel(const bf16* __restrict__ q, const bf16* __restrict__ k,
                 const bf16* __restrict__ vt, const int* __restrict__ seq_id,
                 bf16* __restrict__ ctx) {
  __shared__ __align__(16) bf16 lds_p[8][2 * 16 * 32];   // per-wave P staging
  const int wave = threadIdx.x >> 5, lane = threadIdx.x & 31;
  const int half = lane >> 4, nn = lane & 15;
  const int gw = blockIdx.x * 8 + wave;           // 0..2047
  const int qt = gw & 63, h = (gw >> 6) & 15, b = gw >> 10;
  const int q0 = qt * 32;
  const long bh = (long)(b * H_ + h);

  // Q A-fragments (two 16-row sub-tiles), reused across the whole kv loop
  FragB fq[2][2];
  int sid_q[2][8];
  #pragma unroll
  for (int u = 0; u < 2; u++) {
    const bf16* qrow = q + (bh * L_ + q0 + u * 16 + nn) * DH_;
    #pragma unroll
    for (int c = 0; c < 2; c++) {
      fq[u][c].u[0] = *(const uint4*)(qrow + 32 * c + 16 * half);
      fq[u][c].u[1] = *(const uint4*)(qrow + 32 * c + 16 * half + 8);
    }
    #pragma unroll
    for (int rr = 0; rr < 8; rr++)
      sid_q[u][rr] = seq_id[b * L_ + q0 + u * 16 + 8 * half + rr];
  }

  float m_r[2][8], l_r[2][8];
  #pragma unroll
  for (int u = 0; u < 2; u++)
    #pragma unroll
    for (int rr = 0; rr < 8; rr++) { m_r[u][rr] = -1e30f; l_r[u][rr] = 0.f; }
  v8f acc[2][4] = {};
  bf16* myp = &lds_p[wave][0];

  for (int kv0 = 0; kv0 < L_; kv0 += 32) {
    // ---- scores: each K fragment used by both Q sub-tiles -----------------
    v8f S[2][2] = {};
    #pragma unroll
    for (int t = 0; t < 2; t++) {
      const bf16* krow = k + (bh * L_ + kv0 + 16 * t + nn) * DH_;
      #pragma unroll
      for (int c = 0; c < 2; c++) {
        FragB fk;                      // B frag: n = token, K = Dh dims
        fk.u[0] = *(const uint4*)(krow + 32 * c + 16 * half);
        fk.u[1] = *(const uint4*)(krow + 32 * c + 16 * half + 8);
        #pragma unroll
        for (int u = 0; u < 2; u++)
          S[u][t] = __builtin_amdgcn_wmma_f32_16x16x32_bf16(
              false, fq[u][c].v, false, fk.v, (short)0, S[u][t], false, false);
      }
    }
    const int sid_k0 = seq_id[b * L_ + kv0 + nn];
    const int sid_k1 = seq_id[b * L_ + kv0 + 16 + nn];

    // ---- masked online softmax per sub-tile -------------------------------
    #pragma unroll
    for (int u = 0; u < 2; u++) {
      float p0[8], p1[8], rowm[8], alpha[8];
      #pragma unroll
      for (int rr = 0; rr < 8; rr++) {
        float s0 = S[u][0][rr] * 0.125f;      // 1/sqrt(64)
        float s1 = S[u][1][rr] * 0.125f;
        if (sid_k0 != sid_q[u][rr]) s0 = -1e30f;
        if (sid_k1 != sid_q[u][rr]) s1 = -1e30f;
        p0[rr] = s0; p1[rr] = s1;
        rowm[rr] = fmaxf(s0, s1);
      }
      #pragma unroll
      for (int rr = 0; rr < 8; rr++) {
        #pragma unroll
        for (int off = 8; off; off >>= 1)
          rowm[rr] = fmaxf(rowm[rr], __shfl_xor(rowm[rr], off, 16));
      }
      #pragma unroll
      for (int rr = 0; rr < 8; rr++) {
        float nm = fmaxf(m_r[u][rr], rowm[rr]);
        alpha[rr] = __expf(m_r[u][rr] - nm);
        m_r[u][rr] = nm;
        p0[rr] = __expf(p0[rr] - nm);
        p1[rr] = __expf(p1[rr] - nm);
      }
      #pragma unroll
      for (int rr = 0; rr < 8; rr++) {
        float rs = p0[rr] + p1[rr];
        #pragma unroll
        for (int off = 8; off; off >>= 1) rs += __shfl_xor(rs, off, 16);
        l_r[u][rr] = l_r[u][rr] * alpha[rr] + rs;
      }
      #pragma unroll
      for (int j = 0; j < 4; j++)
        #pragma unroll
        for (int rr = 0; rr < 8; rr++)
          acc[u][j][rr] *= alpha[rr];
      // stage P (16x32) row-major; same-wave LDS ops are in-order (DScnt)
      #pragma unroll
      for (int rr = 0; rr < 8; rr++) {
        myp[u * 512 + (8 * half + rr) * 32 + nn]      = f2bf(p0[rr]);
        myp[u * 512 + (8 * half + rr) * 32 + 16 + nn] = f2bf(p1[rr]);
      }
    }

    // ---- ctx: each V fragment used by both Q sub-tiles --------------------
    FragB fp[2];
    #pragma unroll
    for (int u = 0; u < 2; u++) {
      fp[u].u[0] = *(const uint4*)(myp + u * 512 + nn * 32 + 16 * half);
      fp[u].u[1] = *(const uint4*)(myp + u * 512 + nn * 32 + 16 * half + 8);
    }
    #pragma unroll
    for (int j = 0; j < 4; j++) {
      const bf16* vrow = vt + (bh * DH_ + j * 16 + nn) * L_ + kv0 + 16 * half;
      FragB fv;                        // B frag: n = Dh dim, K = kv tokens
      fv.u[0] = *(const uint4*)(vrow);
      fv.u[1] = *(const uint4*)(vrow + 8);
      #pragma unroll
      for (int u = 0; u < 2; u++)
        acc[u][j] = __builtin_amdgcn_wmma_f32_16x16x32_bf16(
            false, fp[u].v, false, fv.v, (short)0, acc[u][j], false, false);
    }
  }

  #pragma unroll
  for (int u = 0; u < 2; u++)
    #pragma unroll
    for (int j = 0; j < 4; j++)
      #pragma unroll
      for (int rr = 0; rr < 8; rr++) {
        float inv = 1.0f / fmaxf(l_r[u][rr], 1e-20f);
        int row = q0 + u * 16 + 8 * half + rr;
        int col = h * DH_ + j * 16 + nn;
        ctx[((long)(b * L_ + row)) * D_ + col] = f2bf(acc[u][j][rr] * inv);
      }
}

// ---------------------------------------------------------------------------
// launch
// ---------------------------------------------------------------------------
extern "C" void kernel_launch(void* const* d_in, const int* in_sizes, int n_in,
                              void* d_out, int out_size, void* d_ws, size_t ws_size,
                              hipStream_t stream) {
  (void)in_sizes; (void)n_in; (void)out_size; (void)ws_size;
  const float* x    = (const float*)d_in[0];
  const int*   seq  = (const int*)  d_in[1];
  const float* ln1w = (const float*)d_in[2];
  const float* ln1b = (const float*)d_in[3];
  const float* wqkv = (const float*)d_in[4];
  const float* qlnw = (const float*)d_in[5];
  const float* klnw = (const float*)d_in[6];
  const float* wout = (const float*)d_in[7];
  float* out = (float*)d_out;

  char* ws = (char*)d_ws;
  const size_t MB = 1024 * 1024;
  bf16*  h_bf   = (bf16*) (ws + 0);        //  8 MiB  [4096][1024] bf16
  bf16*  wqkv_t = (bf16*) (ws + 8  * MB);  //  6 MiB  [3072][1024] bf16
  bf16*  wout_t = (bf16*) (ws + 14 * MB);  //  2 MiB  [1024][1024] bf16
  float* qkv    = (float*)(ws + 16 * MB);  // 48 MiB  [4096][3072] f32
  bf16*  q_bf   = (bf16*) (ws + 64 * MB);  //  8 MiB  [B][H][L][64]
  bf16*  k_bf   = (bf16*) (ws + 72 * MB);  //  8 MiB  [B][H][L][64]
  bf16*  vt_bf  = (bf16*) (ws + 80 * MB);  //  8 MiB  [B][H][64][L]
  bf16*  ctx_bf = (bf16*) (ws + 88 * MB);  //  8 MiB  [4096][1024]

  cvt_transpose_kernel<<<(D_ * QKVN_) / 256, 256, 0, stream>>>(wqkv, wqkv_t, D_, QKVN_);
  cvt_transpose_kernel<<<(D_ * D_) / 256, 256, 0, stream>>>(wout, wout_t, D_, D_);
  ln1_kernel<<<ROWS_, 256, 0, stream>>>(x, ln1w, ln1b, h_bf);
  gemm_bf16_wmma_kernel<<<dim3(QKVN_ / 128, ROWS_ / 128), 256, 0, stream>>>(
      h_bf, wqkv_t, qkv, ROWS_, QKVN_, D_);
  qk_ln_rope_kernel<<<ROWS_, 256, 0, stream>>>(qkv, qlnw, klnw, q_bf, k_bf, vt_bf);
  attn_kernel<<<(B_ * H_ * (L_ / 32)) / 8, 256, 0, stream>>>(q_bf, k_bf, vt_bf, seq, ctx_bf);
  gemm_bf16_wmma_kernel<<<dim3(D_ / 128, ROWS_ / 128), 256, 0, stream>>>(
      ctx_bf, wout_t, out, ROWS_, D_, D_);
}
